// GPT2Attention_38654705664173
// MI455X (gfx1250) — matrix operand order repaired and checked by reading, not compiled, not run on hardware
//
#include <hip/hip_runtime.h>
#include <stdint.h>

// Problem constants (match reference)
#define B_    2
#define T_    2048
#define C_    1024
#define H_    16
#define HD_   64
#define M_TOT (B_ * T_)   // 4096
#define N_QKV (3 * C_)    // 3072

typedef unsigned short u16;
typedef __attribute__((ext_vector_type(16))) __bf16       v16bf;
typedef __attribute__((ext_vector_type(8)))  float        v8f;
typedef __attribute__((ext_vector_type(4)))  unsigned int u32x4;
typedef __attribute__((ext_vector_type(8)))  int          i32x8;
typedef __attribute__((ext_vector_type(4)))  int          i32x4;

#if __has_builtin(__builtin_amdgcn_tensor_load_to_lds)
#define CDNA5_TDM 1
#else
#define CDNA5_TDM 0
#endif

// ---------- helpers ----------

// fp32 -> bf16, round-to-nearest-even
__device__ __forceinline__ u16 f2bf(float f) {
  unsigned int u = __float_as_uint(f);
  u += 0x7fffu + ((u >> 16) & 1u);
  return (u16)(u >> 16);
}

union Frag16 { u32x4 q[2]; v16bf v; };

// Loads 16 bf16 elements for a WMMA fragment as two 16-byte chunks:
// chunk0 at p, chunk1 at p + elems_apart (in halfword units).
__device__ __forceinline__ v16bf load_frag(const u16* p, int elems_apart) {
  Frag16 f;
  f.q[0] = *(const u32x4*)(p);
  f.q[1] = *(const u32x4*)(p + elems_apart);
  return f.v;
}

__device__ __forceinline__ v8f wmma_bf16(v16bf a, v16bf b, v8f c) {
  // D = A(16x32 bf16) x B(32x16 bf16) + C(16x16 f32)
  return __builtin_amdgcn_wmma_f32_16x16x32_bf16(
      /*neg_a=*/false, a, /*neg_b=*/false, b,
      /*c_mod=*/(short)0, c, /*reuse_a=*/false, /*reuse_b=*/false);
}

__device__ __forceinline__ void wait_tensorcnt0() {
#if __has_builtin(__builtin_amdgcn_s_wait_tensorcnt)
  __builtin_amdgcn_s_wait_tensorcnt(0);
#else
  asm volatile("s_wait_tensorcnt 0" ::: "memory");
#endif
}

#if CDNA5_TDM
// Issue a TDM 2-D tile load (bf16 elements) Global -> LDS via D# descriptor.
// tile_w elements per row, tile_h rows; source rows are row_stride elements apart.
// D# field packing per CDNA5 ISA.
__device__ __forceinline__ void tdm_load_tile_2d(
    const u16* gsrc, u16* ldst,
    unsigned tile_w, unsigned tile_h,
    unsigned tensor_w, unsigned tensor_h, unsigned row_stride) {
  union { unsigned int d[4]; u32x4 v; } g0 = {};
  union { unsigned int d[8]; i32x8 v; } g1 = {};
  union { unsigned int d[4]; i32x4 v; } z4 = {};
  unsigned long long ga = (unsigned long long)gsrc;
  // Flat LDS addresses map to LDS space via addr[31:0] (aperture truncation).
  unsigned int la = (unsigned int)(unsigned long long)ldst;
  // group0: count=1 | lds_addr | global_addr[56:0] | type=2
  g0.d[0] = 1u;
  g0.d[1] = la;
  g0.d[2] = (unsigned int)ga;
  g0.d[3] = (unsigned int)((ga >> 32) & 0x01FFFFFFull) | (2u << 30);
  // group1: data_size=1 (2B) @[17:16]; tensor_dim0 @[79:48]; tensor_dim1 @[111:80];
  //         tile_dim0 @[127:112]; tile_dim1 @[143:128]; tensor_dim0_stride @[207:160]
  g1.d[0] = 1u << 16;
  g1.d[1] = (tensor_w & 0xFFFFu) << 16;
  g1.d[2] = ((tensor_w >> 16) & 0xFFFFu) | ((tensor_h & 0xFFFFu) << 16);
  g1.d[3] = ((tensor_h >> 16) & 0xFFFFu) | (tile_w << 16);
  g1.d[4] = tile_h;                 // tile_dim1 (tile_dim2 = 0)
  g1.d[5] = row_stride;             // tensor_dim0_stride[31:0]
  g1.d[6] = 0;
  g1.d[7] = 0;
#if __has_include(<hip/amd_detail/amd_gfx1250_TDM.h>)
  // amdgpu-toolchain (clang-23 / therock-10.0): 6-arg form
  union { int d[8]; i32x8 v; } z8 = {};
  __builtin_amdgcn_tensor_load_to_lds(g0.v, g1.v, z4.v, z4.v, z8.v, 0);
#else
  // ROCm 7.2 (clang-22): 5-arg form
  __builtin_amdgcn_tensor_load_to_lds(g0.v, g1.v, z4.v, z4.v, 0);
#endif
}
#endif  // CDNA5_TDM

// ---------- cast / transpose kernels ----------

__global__ void cast_f32_bf16_kernel(const float* __restrict__ in,
                                     u16* __restrict__ out, int n) {
  int i = blockIdx.x * blockDim.x + threadIdx.x;
  int stride = gridDim.x * blockDim.x;
  for (; i < n; i += stride) out[i] = f2bf(in[i]);
}

// in: [rows_in][cols_in] f32 -> out: [cols_in][rows_in] bf16 (transposed)
__global__ void transpose_cast_kernel(const float* __restrict__ in,
                                      u16* __restrict__ out,
                                      int rows_in, int cols_in) {
  int i = blockIdx.x * blockDim.x + threadIdx.x;
  int total = rows_in * cols_in;
  int stride = gridDim.x * blockDim.x;
  for (; i < total; i += stride) {
    int n = i / rows_in;   // output row  = input col
    int k = i % rows_in;   // output col  = input row
    out[i] = f2bf(in[(long)k * cols_in + n]);
  }
}

// ---------- shared 128x128 GEMM tile engine ----------
// C_tile(128x128 f32) = A[m0:128][K] * Bt[n0:128][K]^T  (Bt stores B transposed).
// 256 threads = 8 waves; wave w owns rows m0+16w..+15; acc[ns] = 16x16 tile at
// columns n0+16*ns. With TDM: double-buffered DMA of 128x32 tiles into LDS,
// wave 0 issues tensor_load_to_lds for tile k+1 while all waves run WMMAs on
// tile k (TENSORcnt + workgroup barrier for hand-off).
__device__ __forceinline__ void gemm_tile_128x128(
    const u16* __restrict__ A, int lda,
    const u16* __restrict__ Bt, int ldb, int Kdim,
    int m0, int n0, u16* As, u16* Bs, v8f acc[8]) {
  const int tid  = threadIdx.x;
  const int lane = tid & 31, wave = tid >> 5;
  const int half = lane >> 4, ln = lane & 15;
  const v8f zero = {0.f, 0.f, 0.f, 0.f, 0.f, 0.f, 0.f, 0.f};
  for (int i = 0; i < 8; i++) acc[i] = zero;
  const int nk = Kdim / 32;

#if CDNA5_TDM
  if (wave == 0) {
    tdm_load_tile_2d(&A [(long)m0 * lda], As, 32, 128, lda, 128, lda);
    tdm_load_tile_2d(&Bt[(long)n0 * ldb], Bs, 32, 128, ldb, 128, ldb);
  }
  for (int kt = 0; kt < nk; kt++) {
    const int cur = kt & 1;
    if (wave == 0) wait_tensorcnt0();   // DMA into buf[cur] complete
    __syncthreads();                    // publish to all waves
    if (wave == 0 && (kt + 1 < nk)) {   // overlap: DMA next tile into buf[cur^1]
      const int nxt = cur ^ 1;
      tdm_load_tile_2d(&A [(long)m0 * lda + (kt + 1) * 32], &As[nxt * 4096],
                       32, 128, lda, 128, lda);
      tdm_load_tile_2d(&Bt[(long)n0 * ldb + (kt + 1) * 32], &Bs[nxt * 4096],
                       32, 128, ldb, 128, ldb);
    }
    const u16* as = &As[cur * 4096];
    const u16* bs = &Bs[cur * 4096];
    v16bf a = load_frag(&as[(wave * 16 + ln) * 32 + half * 8], 16);
    for (int ns = 0; ns < 8; ns++) {
      v16bf b = load_frag(&bs[(ns * 16 + ln) * 32 + half * 8], 16);
      acc[ns] = wmma_bf16(a, b, acc[ns]);
    }
  }
#else
  for (int kt = 0; kt < nk; kt++) {
    const int k0 = kt * 32;
    {
      int r = tid >> 1;
      int c = (tid & 1) * 16;
      *(u32x4*)&As[r * 32 + c]     = *(const u32x4*)&A [(long)(m0 + r) * lda + k0 + c];
      *(u32x4*)&As[r * 32 + c + 8] = *(const u32x4*)&A [(long)(m0 + r) * lda + k0 + c + 8];
      *(u32x4*)&Bs[r * 32 + c]     = *(const u32x4*)&Bt[(long)(n0 + r) * ldb + k0 + c];
      *(u32x4*)&Bs[r * 32 + c + 8] = *(const u32x4*)&Bt[(long)(n0 + r) * ldb + k0 + c + 8];
    }
    __syncthreads();
    v16bf a = load_frag(&As[(wave * 16 + ln) * 32 + half * 8], 16);
    for (int ns = 0; ns < 8; ns++) {
      v16bf b = load_frag(&Bs[(ns * 16 + ln) * 32 + half * 8], 16);
      acc[ns] = wmma_bf16(a, b, acc[ns]);
    }
    __syncthreads();
  }
#endif
}

// ---------- kernel 1: QKV = Xb @ Wqkv + bqkv, scattered to Q/K/Vt ----------
__global__ __launch_bounds__(256) void qkv_gemm_kernel(
    const u16* __restrict__ Xb, const u16* __restrict__ WqkvT,
    const float* __restrict__ bqkv,
    u16* __restrict__ Qb, u16* __restrict__ Kb, u16* __restrict__ Vtb) {
  __shared__ u16 As[2 * 128 * 32];
  __shared__ u16 Bs[2 * 128 * 32];
  const int n0 = blockIdx.x * 128;
  const int m0 = blockIdx.y * 128;
  v8f acc[8];
  gemm_tile_128x128(Xb, C_, WqkvT, C_, C_, m0, n0, As, Bs, acc);

  const int lane = threadIdx.x & 31, wave = threadIdx.x >> 5;
  const int half = lane >> 4, ln = lane & 15;
  for (int ns = 0; ns < 8; ns++) {
    int col = n0 + ns * 16 + ln;        // 0..3071
    int sec = col / C_;                 // 0=q 1=k 2=v
    int c   = col % C_;
    int h = c / HD_, d = c % HD_;
    float bias = bqkv[col];
    for (int j = 0; j < 8; j++) {
      int row  = m0 + wave * 16 + (half ? j + 8 : j);  // 0..4095
      int bidx = row / T_, t = row % T_;
      int bh   = bidx * H_ + h;
      u16 val  = f2bf(acc[ns][j] + bias);
      if (sec == 0)      Qb [((long)bh * T_  + t) * HD_ + d] = val;  // [BH][T][HD]
      else if (sec == 1) Kb [((long)bh * T_  + t) * HD_ + d] = val;  // [BH][T][HD]
      else               Vtb[((long)bh * HD_ + d) * T_  + t] = val;  // [BH][HD][T]
    }
  }
}

// ---------- kernel 2: causal flash attention ----------
// One wave per 16-row query tile; 4 independent waves per block (no barriers).
__global__ __launch_bounds__(128) void attn_kernel(
    const u16* __restrict__ Qb, const u16* __restrict__ Kb,
    const u16* __restrict__ Vtb, u16* __restrict__ Yb) {
  __shared__ u16 Ps[4][16 * 32];   // per-wave P staging tile (16 q x 32 keys, bf16)
  const int tid  = threadIdx.x;
  const int lane = tid & 31, wave = tid >> 5;
  const int half = lane >> 4, ln = lane & 15;
  const int QT = T_ / 16;                 // 128 q-tiles per (b,h)
  const int gq = blockIdx.x * 4 + wave;   // global q-tile id
  const int bh = gq / QT;
  const int qt = gq % QT;
  const u16* Qp = Qb  + (long)bh * T_  * HD_;
  const u16* Kp = Kb  + (long)bh * T_  * HD_;
  const u16* Vp = Vtb + (long)bh * HD_ * T_;
  u16* ps = Ps[wave];

  // Q fragments for both 32-wide K-steps covering HD=64 (persist in VGPRs)
  v16bf qf[2];
  for (int ks = 0; ks < 2; ks++)
    qf[ks] = load_frag(Qp + (long)(qt * 16 + ln) * HD_ + ks * 32 + half * 8, 16);

  const v8f zero = {0.f, 0.f, 0.f, 0.f, 0.f, 0.f, 0.f, 0.f};
  float mrow[8], lrow[8];
  v8f acc[4];                              // output 16 x 64 (4 d-subtiles)
  for (int j = 0; j < 8; j++) { mrow[j] = -1e30f; lrow[j] = 0.f; }
  for (int i = 0; i < 4; i++) acc[i] = zero;

  const int   nkeys = (qt + 1) * 16;       // causal: keys 0..nkeys-1
  const int   nkb   = (nkeys + 31) / 32;   // 32-key blocks
  const float scale = 0.125f;              // 1/sqrt(64)

  for (int kb = 0; kb < nkb; kb++) {
    const int key0 = kb * 32;
    if (kb + 1 < nkb) {
      // gfx1250 global_prefetch_b8 hints for the next 32-key block
      __builtin_prefetch(Kp + (long)(key0 + 32 + lane) * HD_, 0, 0);
      __builtin_prefetch(Vp + (long)lane * T_ + key0 + 32, 0, 0);
      __builtin_prefetch(Vp + (long)(32 + lane) * T_ + key0 + 32, 0, 0);
    }
    // ---- S = Q K^T for 32 keys (two 16x16 f32 tiles) ----
    v8f s0 = zero, s1 = zero;
    for (int ks = 0; ks < 2; ks++) {
      v16bf b0 = load_frag(Kp + (long)(key0      + ln) * HD_ + ks * 32 + half * 8, 16);
      v16bf b1 = load_frag(Kp + (long)(key0 + 16 + ln) * HD_ + ks * 32 + half * 8, 16);
      s0 = wmma_bf16(qf[ks], b0, s0);
      s1 = wmma_bf16(qf[ks], b1, s1);
    }
    // ---- scale + causal mask + online softmax (fp32) ----
    float p0[8], p1[8];
    for (int j = 0; j < 8; j++) {
      int   qrow = qt * 16 + (half ? j + 8 : j);
      float a0 = s0[j] * scale, a1 = s1[j] * scale;
      if (key0 + ln      > qrow) a0 = -1e30f;
      if (key0 + 16 + ln > qrow) a1 = -1e30f;
      p0[j] = a0; p1[j] = a1;
    }
    for (int j = 0; j < 8; j++) {
      float rm = fmaxf(p0[j], p1[j]);
      for (int off = 1; off < 16; off <<= 1) rm = fmaxf(rm, __shfl_xor(rm, off, 16));
      float mnew  = fmaxf(mrow[j], rm);
      float alpha = __expf(mrow[j] - mnew);
      float e0 = __expf(p0[j] - mnew), e1 = __expf(p1[j] - mnew);
      float rs = e0 + e1;
      for (int off = 1; off < 16; off <<= 1) rs += __shfl_xor(rs, off, 16);
      lrow[j] = lrow[j] * alpha + rs;
      mrow[j] = mnew;
      for (int ds = 0; ds < 4; ds++) acc[ds][j] *= alpha;
      int row = half ? j + 8 : j;          // C-layout row for this VGPR
      ps[row * 32 + ln]      = f2bf(e0);   // keys key0+0..15
      ps[row * 32 + 16 + ln] = f2bf(e1);   // keys key0+16..31
    }
    asm volatile("s_wait_dscnt 0" ::: "memory");  // cross-lane LDS visibility
    // ---- PV: A = P(16x32 bf16 from LDS), B = Vt rows (contiguous in key) ----
    v16bf pa = load_frag(&ps[ln * 32 + half * 8], 16);
    for (int ds = 0; ds < 4; ds++) {
      v16bf vb = load_frag(Vp + (long)(ds * 16 + ln) * T_ + key0 + half * 8, 16);
      acc[ds] = wmma_bf16(pa, vb, acc[ds]);
    }
    asm volatile("s_wait_dscnt 0" ::: "memory");  // ps reads done before next overwrite
  }

  // ---- finalize: y = acc / l, write [B][T][H][HD] as bf16 ----
  int b = bh / H_, h = bh % H_;
  for (int j = 0; j < 8; j++) {
    int   q   = qt * 16 + (half ? j + 8 : j);
    float inv = 1.0f / lrow[j];
    for (int ds = 0; ds < 4; ds++) {
      Yb[((long)(b * T_ + q)) * C_ + h * HD_ + ds * 16 + ln] = f2bf(acc[ds][j] * inv);
    }
  }
}

// ---------- kernel 3: out = Y @ Wproj + bproj (fp32 out) ----------
__global__ __launch_bounds__(256) void proj_gemm_kernel(
    const u16* __restrict__ Yb, const u16* __restrict__ WprojT,
    const float* __restrict__ bproj, float* __restrict__ out) {
  __shared__ u16 As[2 * 128 * 32];
  __shared__ u16 Bs[2 * 128 * 32];
  const int n0 = blockIdx.x * 128;
  const int m0 = blockIdx.y * 128;
  v8f acc[8];
  gemm_tile_128x128(Yb, C_, WprojT, C_, C_, m0, n0, As, Bs, acc);

  const int lane = threadIdx.x & 31, wave = threadIdx.x >> 5;
  const int half = lane >> 4, ln = lane & 15;
  for (int ns = 0; ns < 8; ns++) {
    int col = n0 + ns * 16 + ln;
    float bias = bproj[col];
    for (int j = 0; j < 8; j++) {
      int row = m0 + wave * 16 + (half ? j + 8 : j);
      out[(long)row * C_ + col] = acc[ns][j] + bias;
    }
  }
}

// ---------- host launcher ----------
extern "C" void kernel_launch(void* const* d_in, const int* in_sizes, int n_in,
                              void* d_out, int out_size, void* d_ws, size_t ws_size,
                              hipStream_t stream) {
  const float* x     = (const float*)d_in[0];
  const float* Wqkv  = (const float*)d_in[1];
  const float* bqkv  = (const float*)d_in[2];
  const float* Wproj = (const float*)d_in[3];
  const float* bproj = (const float*)d_in[4];
  float* out = (float*)d_out;

  // workspace layout (bf16 buffers), ~48 MiB total
  char* ws = (char*)d_ws;
  u16* Xb     = (u16*)(ws);                     //  8 MiB: x as bf16 [4096][1024]
  u16* WqkvT  = (u16*)(ws + (8ull  << 20));     //  6 MiB: Wqkv^T  [3072][1024]
  u16* WprojT = (u16*)(ws + (14ull << 20));     //  2 MiB: Wproj^T [1024][1024]
  u16* Qb     = (u16*)(ws + (16ull << 20));     //  8 MiB: Q  [BH][T][HD]
  u16* Kb     = (u16*)(ws + (24ull << 20));     //  8 MiB: K  [BH][T][HD]
  u16* Vtb    = (u16*)(ws + (32ull << 20));     //  8 MiB: V^T[BH][HD][T]
  u16* Yb     = (u16*)(ws + (40ull << 20));     //  8 MiB: attn out bf16 [4096][1024]

  hipLaunchKernelGGL(cast_f32_bf16_kernel, dim3(2048), dim3(256), 0, stream,
                     x, Xb, M_TOT * C_);
  hipLaunchKernelGGL(transpose_cast_kernel, dim3(2048), dim3(256), 0, stream,
                     Wqkv, WqkvT, C_, N_QKV);
  hipLaunchKernelGGL(transpose_cast_kernel, dim3(1024), dim3(256), 0, stream,
                     Wproj, WprojT, C_, C_);
  hipLaunchKernelGGL(qkv_gemm_kernel, dim3(N_QKV / 128, M_TOT / 128), dim3(256), 0, stream,
                     Xb, WqkvT, bqkv, Qb, Kb, Vtb);
  hipLaunchKernelGGL(attn_kernel, dim3((B_ * H_ * (T_ / 16)) / 4), dim3(128), 0, stream,
                     Qb, Kb, Vtb, Yb);
  hipLaunchKernelGGL(proj_gemm_kernel, dim3(C_ / 128, M_TOT / 128), dim3(256), 0, stream,
                     Yb, WprojT, bproj, out);
}
